// DotProductAttention_35854386987770
// MI455X (gfx1250) — matrix operand order repaired and checked by reading, not compiled, not run on hardware
//
#include <hip/hip_runtime.h>
#include <hip/hip_bf16.h>

typedef __attribute__((ext_vector_type(16))) __bf16 v16bf;
typedef __attribute__((ext_vector_type(4)))  __bf16 v4bf;
typedef __attribute__((ext_vector_type(8)))  float  v8f;

#define WMMA_BF16(A, B, C) \
    __builtin_amdgcn_wmma_f32_16x16x32_bf16(false, (A), false, (B), (short)0, (C), false, false)

// Scores are computed directly in log2 domain: fold 1/sqrt(64) * log2(e) into Q.
#define QSCALE   (0.125f * 1.44269504088896340736f)
#define MASKVAL  (-1.0e6f * 1.44269504088896340736f)   // reference's -1e6, log2 domain

// ---------------------------------------------------------------------------
// Pre-pass 1: K fp32 -> bf16, same [b][s][d] layout.
// ---------------------------------------------------------------------------
__global__ __launch_bounds__(256)
void prep_k_kernel(const float* __restrict__ K, __bf16* __restrict__ Kbf)
{
    const size_t n4 = (size_t)64 * 1024 * 64 / 4;
    const size_t stride = (size_t)gridDim.x * blockDim.x;
    for (size_t i = (size_t)blockIdx.x * blockDim.x + threadIdx.x; i < n4; i += stride) {
        const float4 f = ((const float4*)K)[i];
        v4bf o;
        o[0] = (__bf16)f.x; o[1] = (__bf16)f.y;
        o[2] = (__bf16)f.z; o[3] = (__bf16)f.w;
        ((v4bf*)Kbf)[i] = o;
    }
}

// ---------------------------------------------------------------------------
// Pre-pass 2: V fp32 [b][s][d] -> bf16 transposed Vt [b][d][s] via LDS tile.
// ---------------------------------------------------------------------------
__global__ __launch_bounds__(256)
void prep_v_kernel(const float* __restrict__ V, __bf16* __restrict__ Vt)
{
    constexpr int S = 1024, D = 64;
    __shared__ float tile[64][65];

    const int b  = blockIdx.x >> 4;
    const int s0 = (blockIdx.x & 15) * 64;
    const int c  = threadIdx.x & 63;
    const int r4 = threadIdx.x >> 6;

    #pragma unroll
    for (int i = 0; i < 16; ++i) {
        const int sl = i * 4 + r4;
        tile[sl][c] = V[((size_t)b * S + s0 + sl) * D + c];
    }
    __syncthreads();
    #pragma unroll
    for (int i = 0; i < 16; ++i) {
        const int dl = i * 4 + r4;
        Vt[((size_t)b * D + dl) * S + s0 + c] = (__bf16)tile[c][dl];
    }
}

// ---------------------------------------------------------------------------
// Flash attention: B=64, S=1024, D=64, fp32 in/out, bf16 WMMA compute.
// One wave = 16 queries x full D. One block = 8 waves = 128 queries.
// ---------------------------------------------------------------------------
__global__ __launch_bounds__(256)
void fattn_gfx1250_kernel(const float*  __restrict__ Q,
                          const __bf16* __restrict__ Kbf,   // [b][s][d]
                          const __bf16* __restrict__ Vt,    // [b][d][s]
                          const int*    __restrict__ VL,
                          float*        __restrict__ O)
{
    constexpr int S = 1024, D = 64;
    __shared__ float sS[8][16 * 33];   // per-wave C->A score transpose staging

    const int wave = threadIdx.x >> 5;
    const int lane = threadIdx.x & 31;
    const int g    = lane >> 4;
    const int ln   = lane & 15;

    const int b    = blockIdx.x >> 3;
    const int qt   = blockIdx.x & 7;
    const int q0   = qt * 128 + wave * 16;
    const int vlen = VL[b];

    const float*  qb = Q   + (size_t)b * S * D;
    const __bf16* kw = Kbf + (size_t)b * S * D;
    const __bf16* vw = Vt  + (size_t)b * D * S;

    // ---- Q tile (16x64) into WMMA A layout, scaled into log2 domain.
    v16bf qa[2];
    {
        const float* qrow = qb + (size_t)(q0 + ln) * D;
        #pragma unroll
        for (int h = 0; h < 2; ++h) {
            #pragma unroll
            for (int v = 0; v < 8; ++v) {
                const int kk = h * 32 + g * 16 + v * 2;
                qa[h][2 * v]     = (__bf16)(qrow[kk]     * QSCALE);
                qa[h][2 * v + 1] = (__bf16)(qrow[kk + 1] * QSCALE);
            }
        }
    }

    v8f acc0 = {}, acc1 = {}, acc2 = {}, acc3 = {};
    float m_run = -__builtin_inff();
    float l_run = 0.0f;
    float* Sw = sS[wave];

    for (int key0 = 0; key0 < S; key0 += 32) {
        if (vlen > 0 && key0 >= vlen) break;   // block-uniform early exit

        // ---- Issue ALL tile loads up front (16x b128): K for QK^T now,
        // V for P@V later -- V latency hides under WMMA + softmax.
        const v16bf kOp0 = *(const v16bf*)(kw + (size_t)(key0 + ln)      * D +  0 + g * 16);
        const v16bf kOp1 = *(const v16bf*)(kw + (size_t)(key0 + 16 + ln) * D +  0 + g * 16);
        const v16bf kOp2 = *(const v16bf*)(kw + (size_t)(key0 + ln)      * D + 32 + g * 16);
        const v16bf kOp3 = *(const v16bf*)(kw + (size_t)(key0 + 16 + ln) * D + 32 + g * 16);
        const v16bf vOp0 = *(const v16bf*)(vw + (size_t)(0  + ln) * S + key0 + g * 16);
        const v16bf vOp1 = *(const v16bf*)(vw + (size_t)(16 + ln) * S + key0 + g * 16);
        const v16bf vOp2 = *(const v16bf*)(vw + (size_t)(32 + ln) * S + key0 + g * 16);
        const v16bf vOp3 = *(const v16bf*)(vw + (size_t)(48 + ln) * S + key0 + g * 16);

        // ---- Scores S(16x32) = Q @ K^T (log2 domain).
        v8f s0 = {}, s1 = {};
        s0 = WMMA_BF16(qa[0], kOp0, s0);
        s1 = WMMA_BF16(qa[0], kOp1, s1);
        s0 = WMMA_BF16(qa[1], kOp2, s0);
        s1 = WMMA_BF16(qa[1], kOp3, s1);

        // ---- C-layout -> A-layout transpose via LDS.
        #pragma unroll
        for (int v = 0; v < 8; ++v) {
            Sw[(g * 8 + v) * 33 + ln]      = s0[v];
            Sw[(g * 8 + v) * 33 + 16 + ln] = s1[v];
        }
        asm volatile("s_wait_dscnt 0x0" ::: "memory");

        float p[16];
        #pragma unroll
        for (int j = 0; j < 8; ++j) p[j]     = Sw[ln * 33 + g * 16 + j];
        #pragma unroll
        for (int j = 0; j < 8; ++j) p[8 + j] = Sw[ln * 33 + g * 16 + 8 + j];

        // ---- Mask only the partial tile (block-uniform branch).
        if (key0 + 32 > vlen) {
            #pragma unroll
            for (int e = 0; e < 16; ++e) {
                const int kg = key0 + g * 16 + e;
                p[e] = (kg < vlen) ? p[e] : MASKVAL;
            }
        }

        float mx = -__builtin_inff();
        #pragma unroll
        for (int e = 0; e < 16; ++e) mx = fmaxf(mx, p[e]);
        mx = fmaxf(mx, __shfl_xor(mx, 16, 32));   // row lives in lane pair (L, L+16)

        // ---- Online softmax. Unconditional rescale: alpha == 1.0 exactly when
        // the row max is unchanged, and branch-free accumulators keep the tied
        // WMMA src2/dst registers coalesced (no per-iteration phi copies).
        const float m_new = fmaxf(m_run, mx);
        const float alpha = __builtin_amdgcn_exp2f(m_run - m_new);
        m_run = m_new;
        #pragma unroll
        for (int v = 0; v < 8; ++v) {
            const float av = __shfl(alpha, g * 8 + v, 32);
            acc0[v] *= av; acc1[v] *= av; acc2[v] *= av; acc3[v] *= av;
        }

        float lsum = 0.0f;
        #pragma unroll
        for (int e = 0; e < 16; ++e) {
            p[e] = __builtin_amdgcn_exp2f(p[e] - m_new);
            lsum += p[e];
        }
        lsum += __shfl_xor(lsum, 16, 32);
        l_run = l_run * alpha + lsum;

        // ---- Pack P to bf16 (already in A layout); O += P @ V.
        v16bf pa;
        #pragma unroll
        for (int e = 0; e < 16; ++e) pa[e] = (__bf16)p[e];

        acc0 = WMMA_BF16(pa, vOp0, acc0);
        acc1 = WMMA_BF16(pa, vOp1, acc1);
        acc2 = WMMA_BF16(pa, vOp2, acc2);
        acc3 = WMMA_BF16(pa, vOp3, acc3);
    }

    // ---- Finalize: per-row 1/l (rcp) and coalesced stores.
    #pragma unroll
    for (int v = 0; v < 8; ++v) {
        const float lr  = __shfl(l_run, g * 8 + v, 32);
        const float inv = __builtin_amdgcn_rcpf(lr);
        float* orow = O + ((size_t)b * S + q0 + g * 8 + v) * D + ln;
        orow[0]  = acc0[v] * inv;
        orow[16] = acc1[v] * inv;
        orow[32] = acc2[v] * inv;
        orow[48] = acc3[v] * inv;
    }
}

extern "C" void kernel_launch(void* const* d_in, const int* in_sizes, int n_in,
                              void* d_out, int out_size, void* d_ws, size_t ws_size,
                              hipStream_t stream) {
    const float* q  = (const float*)d_in[0];
    const float* k  = (const float*)d_in[1];
    const float* v  = (const float*)d_in[2];
    const int*   vl = (const int*)d_in[3];
    float* out = (float*)d_out;

    // Workspace layout: Kbf (8 MB) | Vt (8 MB)  -- bf16, 64*1024*64 each.
    __bf16* kbf = (__bf16*)d_ws;
    __bf16* vt  = kbf + (size_t)64 * 1024 * 64;

    prep_k_kernel<<<dim3(2048), dim3(256), 0, stream>>>(k, kbf);
    prep_v_kernel<<<dim3(64 * 16), dim3(256), 0, stream>>>(v, vt);

    fattn_gfx1250_kernel<<<dim3(64 * 8), dim3(256), 0, stream>>>(q, kbf, vt, vl, out);
}
